// PPNUResNet_44590350467187
// MI455X (gfx1250) — matrix operand order repaired and checked by reading, not compiled, not run on hardware
//
#include <hip/hip_runtime.h>

typedef _Float16 half_t;
typedef __attribute__((ext_vector_type(16))) _Float16 v16h;
typedef __attribute__((ext_vector_type(8)))  _Float16 v8h;
typedef __attribute__((ext_vector_type(8)))  float    v8f;

#define TPB 256

// ---------------------------------------------------------------------------
// Weight repack: f32 DHWIO -> f16 WMMA B-fragment order, zero padded.
// wr[(((t*KC+kc)*CT+ct)*32+lane)*16+e] = w[t][kc*32+kk(hi,e)][ct*16+ln]
//   (B 32x16 f16 layout: column = lane&15, K = e<8 ? hi*8+e : 16+hi*8+(e-8))
// ---------------------------------------------------------------------------
__global__ void wrepack_k(const float* __restrict__ w, half_t* __restrict__ wr,
                          int taps, int CinW, int Cout, int KC, int CT) {
    long total = (long)taps * KC * CT * 512;
    long i = (long)blockIdx.x * blockDim.x + threadIdx.x;
    if (i >= total) return;
    int e = (int)(i & 15); long r = i >> 4;
    int lane = (int)(r & 31); r >>= 5;
    int ct = (int)(r % CT); r /= CT;
    int kc = (int)(r % KC); int t = (int)(r / KC);
    int hi = lane >> 4, ln = lane & 15;
    int k = kc * 32 + ((e < 8) ? (hi * 8 + e) : (16 + hi * 8 + (e - 8)));
    int n = ct * 16 + ln;
    float v = (k < CinW && n < Cout) ? w[((long)t * CinW + k) * Cout + n] : 0.f;
    wr[i] = (half_t)v;
}

// ---------------------------------------------------------------------------
// Implicit-GEMM conv on V_WMMA_F32_16X16X32_F16.
// MODE 0: 3x3x3 SAME | 1: 2x2x2 s2 down | 2: 2x2x2 s2 transpose (parity tiled)
// MODE 3: 1x1x1. Wave tile: M=16 voxels x N=32 out-channels (2 acc chains).
// A: two aligned b128 loads/lane/chunk; B: one aligned v16h load per tile.
// Epilogue fuses mask multiply, residual add, optional f16 mirror store.
// ---------------------------------------------------------------------------
template<int MODE>
__global__ void __launch_bounds__(TPB) conv_wmma(
    const half_t* __restrict__ in, const half_t* __restrict__ wt,
    const float* __restrict__ mask, const float* __restrict__ addsrc,
    float* __restrict__ out, half_t* __restrict__ outh,
    int H, int W, int inD, int inH, int inW,
    int Cin, int Cout, int nWaves)
{
    int wid = blockIdx.x * (blockDim.x >> 5) + (threadIdx.x >> 5);
    if (wid >= nWaves) return;
    int lane = threadIdx.x & 31;
    int hi = lane >> 4, ln = lane & 15;

    int CT = (Cout + 15) >> 4;
    int CTP = (CT + 1) >> 1;
    int KC = (Cin + 31) >> 5;
    int mt = (MODE == 2) ? (((inW + 15) >> 4) << 1) : ((W + 15) >> 4);
    int rowT = mt * CTP;
    int zy = wid / rowT;  int r = wid - zy * rowT;
    int ct2 = r / mt;     int xt = r - ct2 * mt;
    int z0 = zy / H;      int y0 = zy - z0 * H;
    int hasT1 = __builtin_amdgcn_readfirstlane(((ct2 << 1) + 1) < CT ? 1 : 0);

    // A-row voxel (m = ln) output x coordinate
    int xo = 0, cx = 0, xi = 0;
    bool mvalid;
    if (MODE == 2) { cx = xt & 1; xi = ((xt >> 1) << 4) + ln; xo = (xi << 1) + cx;
                     mvalid = (xi < inW) && (xo < W); }
    else           { xo = (xt << 4) + ln; mvalid = xo < W; }

    constexpr int ntaps = (MODE == 0) ? 27 : (MODE == 1) ? 8 : 1;
    v8f acc0 = {}, acc1 = {};
    const v8h z8 = {};

    #pragma unroll
    for (int t = 0; t < ntaps; ++t) {
        int iz, iy, ix, tw = t;
        if (MODE == 0)      { iz = z0 + t / 9 - 1; iy = y0 + (t / 3) % 3 - 1; ix = xo + t % 3 - 1; }
        else if (MODE == 1) { iz = (z0 << 1) + (t >> 2); iy = (y0 << 1) + ((t >> 1) & 1); ix = (xo << 1) + (t & 1); }
        else if (MODE == 2) { iz = z0 >> 1; iy = y0 >> 1; ix = xi; tw = ((z0 & 1) << 2) | ((y0 & 1) << 1) | cx; }
        else                { iz = z0; iy = y0; ix = xo; }
        bool ok = mvalid && iz >= 0 && iz < inD && iy >= 0 && iy < inH && ix >= 0 && ix < inW;
        const half_t* ap = in + (((long)iz * inH + iy) * inW + ix) * Cin + hi * 8;
        const half_t* wp = wt + (long)tw * KC * CT * 512;

        for (int kc = 0; kc < KC; ++kc) {
            // A fragment: two contiguous 8-half runs per lane (16B aligned)
            const half_t* ab = ap + kc * 32;
            v8h a0 = ok ? *(const v8h*)(ab)      : z8;
            v8h a1 = ok ? *(const v8h*)(ab + 16) : z8;
            v16h A = __builtin_shufflevector(a0, a1,
                     0, 1, 2, 3, 4, 5, 6, 7, 8, 9, 10, 11, 12, 13, 14, 15);
            // B fragments: pre-repacked, contiguous, unguarded
            const half_t* bb = wp + ((long)kc * CT + (ct2 << 1)) * 512 + (lane << 4);
            acc0 = __builtin_amdgcn_wmma_f32_16x16x32_f16(
                false, A, false, *(const v16h*)bb, (short)0, acc0, false, false);
            if (hasT1)
                acc1 = __builtin_amdgcn_wmma_f32_16x16x32_f16(
                    false, A, false, *(const v16h*)(bb + 512), (short)0, acc1, false, false);
        }
    }

    // Epilogue: D layout VGPR rr -> M = rr + hi*8, N = lane&15
    #pragma unroll
    for (int ti = 0; ti < 2; ++ti) {
        if (ti == 1 && !hasT1) break;
        int n = (ct2 << 5) + (ti << 4) + ln;
        if (n < Cout) {
            #pragma unroll
            for (int rr = 0; rr < 8; ++rr) {
                int m = rr + (hi << 3);
                int xs; bool ok2;
                if (MODE == 2) { int xi2 = ((xt >> 1) << 4) + m; xs = (xi2 << 1) + cx;
                                 ok2 = (xi2 < inW) && (xs < W); }
                else           { xs = (xt << 4) + m; ok2 = xs < W; }
                if (ok2) {
                    long vox = ((long)z0 * H + y0) * W + xs;
                    float val = (ti == 0) ? acc0[rr] : acc1[rr];
                    if (mask)   val *= mask[vox];
                    if (addsrc) val += addsrc[vox * Cout + n];
                    if (out)    out[vox * Cout + n] = val;
                    if (outh)   outh[vox * Cout + n] = (half_t)val;
                }
            }
        }
    }
}

// ---------------------------------------------------------------------------
// Elementwise / reduction helper kernels
// ---------------------------------------------------------------------------
__global__ void scatter_k(const float* __restrict__ pc, float* grid, float* occ, int n) {
    int i = blockIdx.x * blockDim.x + threadIdx.x; if (i >= n) return;
    int ix = (int)pc[i * 5 + 0], iy = (int)pc[i * 5 + 1], iz = (int)pc[i * 5 + 2];
    float f = pc[i * 5 + 4];
    long v = ((long)ix * 96 + iy) * 96 + iz;
    atomicAdd(grid + v, f);
    occ[v] = 1.0f;
}

__global__ void pool_k(const float* __restrict__ m, float* o, int d) {
    int i = blockIdx.x * blockDim.x + threadIdx.x;
    int V = d * d * d; if (i >= V) return;
    int z = i / (d * d), rem = i - z * d * d, y = rem / d, x = rem - y * d;
    int id = 2 * d; float best = -1e30f;
    for (int a = 0; a < 2; ++a) for (int b = 0; b < 2; ++b) for (int c = 0; c < 2; ++c)
        best = fmaxf(best, m[(((long)(2 * z + a) * id + 2 * y + b) * id) + 2 * x + c]);
    o[i] = best;
}

__global__ void cast_k(const float* __restrict__ a, half_t* b, long n) {
    long i = (long)blockIdx.x * blockDim.x + threadIdx.x;
    if (i < n) b[i] = (half_t)a[i];
}

__global__ void expand16_k(const float* __restrict__ g, half_t* o, long V) {
    long i = (long)blockIdx.x * blockDim.x + threadIdx.x;
    if (i >= V * 16) return;
    long v = i >> 4; int c = (int)(i & 15);
    o[i] = (half_t)((c == 0) ? g[v] : 0.f);
}

__global__ void bn_stats(const float* __restrict__ f, const float* __restrict__ mk,
                         float* __restrict__ st, long V, int C) {
    int c = blockIdx.y;
    __shared__ float sA[TPB], sB[TPB], sC[TPB];
    float a = 0.f, b = 0.f, cn = 0.f;
    for (long v = (long)blockIdx.x * blockDim.x + threadIdx.x; v < V;
         v += (long)gridDim.x * blockDim.x) {
        float m = mk[v]; float x = f[v * C + c] * m;   // m in {0,1} -> m^2 = m
        a += x; b += x * x; cn += m;
    }
    sA[threadIdx.x] = a; sB[threadIdx.x] = b; sC[threadIdx.x] = cn;
    __syncthreads();
    for (int s = TPB / 2; s > 0; s >>= 1) {
        if ((int)threadIdx.x < s) {
            sA[threadIdx.x] += sA[threadIdx.x + s];
            sB[threadIdx.x] += sB[threadIdx.x + s];
            sC[threadIdx.x] += sC[threadIdx.x + s];
        }
        __syncthreads();
    }
    if (threadIdx.x == 0) {
        atomicAdd(st + c, sA[0]);
        atomicAdd(st + C + c, sB[0]);
        if (c == 0) atomicAdd(st + 2 * C, sC[0]);
    }
}

__global__ void bn_apply(const float* __restrict__ f, const float* __restrict__ mk,
                         const float* __restrict__ st, const float* __restrict__ sc,
                         const float* __restrict__ bi, half_t* __restrict__ oh,
                         long V, int C) {
    long i = (long)blockIdx.x * blockDim.x + threadIdx.x;
    if (i >= V * C) return;
    long v = i / C; int c = (int)(i - v * C);
    float n = fmaxf(st[2 * C], 1.0f);
    float mean = st[c] / n;
    float var = st[C + c] / n - mean * mean;
    float h = (f[i] - mean) * rsqrtf(var + 1e-4f) * sc[c] + bi[c];
    oh[i] = (half_t)(fmaxf(h, 0.f) * mk[v]);
}

__global__ void concat_k(const float* __restrict__ a, const float* __restrict__ b,
                         float* o, long V, int C) {
    long i = (long)blockIdx.x * blockDim.x + threadIdx.x;
    if (i >= V * 2 * C) return;
    long v = i / (2 * C); int c = (int)(i - v * 2 * C);
    o[i] = (c < C) ? a[v * C + c] : b[v * C + (c - C)];
}

__global__ void att_k(const float* __restrict__ s, const float* __restrict__ m,
                      float* att, int V) {
    int i = blockIdx.x * blockDim.x + threadIdx.x; if (i >= V) return;
    float p = 1.f / (1.f + expf(s[i * 2] - s[i * 2 + 1]));  // softmax[...,1]
    att[i] = ((p > 0.8f) ? 1.f : 0.f) * m[i];
}

__global__ void upwrite_k(const float* __restrict__ att, float* o, int dc, int df) {
    int i = blockIdx.x * blockDim.x + threadIdx.x;
    int V = df * df * df; if (i >= V) return;
    int z = i / (df * df), rem = i - z * df * df, y = rem / df, x = rem - y * df;
    int f = df / dc;
    o[i] = att[((z / f) * dc + y / f) * dc + x / f];
}

__global__ void mulatt_k(const float* __restrict__ f, const float* __restrict__ att,
                         half_t* o, int d, int dc, int C) {
    long i = (long)blockIdx.x * blockDim.x + threadIdx.x;
    long V = (long)d * d * d; if (i >= V * C) return;
    long v = i / C;
    int z = (int)(v / (d * d)); int rem = (int)(v - (long)z * d * d);
    int y = rem / d, x = rem - y * d;
    int s = d / dc;
    o[i] = (half_t)(f[i] * att[((z / s) * dc + y / s) * dc + x / s]);
}

__global__ void seg_k(const float* __restrict__ pc, const half_t* __restrict__ h,
                      const float* __restrict__ lw, const float* __restrict__ lb,
                      float* o, int n) {
    int i = blockIdx.x * blockDim.x + threadIdx.x; if (i >= n) return;
    int ix = (int)pc[i * 5 + 0], iy = (int)pc[i * 5 + 1], iz = (int)pc[i * 5 + 2];
    long vox = (((long)ix * 96 + iy) * 96 + iz) * 16;
    for (int c = 0; c < 5; ++c) {
        float acc = lb[c];
        for (int k = 0; k < 16; ++k) acc += (float)h[vox + k] * lw[k * 5 + c];
        o[i * 5 + c] = acc;
    }
}

__global__ void p3gather_k(const float* __restrict__ pc, const float* __restrict__ pix,
                           const float* __restrict__ sco, float* o, int n) {
    int i = blockIdx.x * blockDim.x + threadIdx.x; if (i >= n) return;
    int ix = (int)pc[i * 5 + 0], iy = (int)pc[i * 5 + 1], iz = (int)pc[i * 5 + 2];
    long vox = ((long)ix * 96 + iy) * 96 + iz;
    o[i * 5 + 0] = pix[vox * 3 + 0];
    o[i * 5 + 1] = pix[vox * 3 + 1];
    o[i * 5 + 2] = pix[vox * 3 + 2];
    o[i * 5 + 3] = sco[vox * 2 + 0];
    o[i * 5 + 4] = sco[vox * 2 + 1];
}

// ---------------------------------------------------------------------------
// Host orchestration
// ---------------------------------------------------------------------------
struct BlkP { const float *bn1s, *bn1b, *bn2s, *bn2b, *nin, *w1, *w2; };
struct EncP { BlkP blk[2]; const float *dbns, *dbnb, *dw; };
struct DecP { BlkP blk[2]; const float *bns, *bnb, *dcw; };

static inline unsigned gsz(long n) { return (unsigned)((n + TPB - 1) / TPB); }

extern "C" void kernel_launch(void* const* d_in, const int* in_sizes, int n_in,
                              void* d_out_v, int out_size, void* d_ws, size_t ws_size,
                              hipStream_t stream) {
    if (n_in < 149) return;
    const float* pc = (const float*)d_in[0];
    const int N = in_sizes[0] / 5;
    float* d_out = (float*)d_out_v;

    // ---- parameter pytree walk (sorted dict keys, jax flatten order) ----
    int pi = 1;
    auto P = [&]() { return (const float*)d_in[pi++]; };
    DecP dec[4];
    for (int i = 0; i < 4; ++i) {                       // 'blocks' < 'bn' < 'deconv'
        for (int b = 0; b < 2; ++b) {                   // bn1, bn2, [nin], w1, w2
            BlkP& k = dec[i].blk[b];
            k.bn1s = P(); k.bn1b = P(); k.bn2s = P(); k.bn2b = P();
            k.nin = (b == 0) ? P() : nullptr;
            k.w1 = P(); k.w2 = P();
        }
        dec[i].bns = P(); dec[i].bnb = P(); dec[i].dcw = P();
    }
    EncP enc[5];
    for (int i = 0; i < 5; ++i) {                       // 'blocks' < 'down_bn' < 'down_w'
        for (int b = 0; b < 2; ++b) {
            BlkP& k = enc[i].blk[b];
            k.bn1s = P(); k.bn1b = P(); k.bn2s = P(); k.bn2b = P();
            k.nin = nullptr; k.w1 = P(); k.w2 = P();
        }
        if (i < 4) { enc[i].dbns = P(); enc[i].dbnb = P(); enc[i].dw = P(); }
    }
    const float* input_conv = P();
    const float* linw = P(); const float* linb = P();
    const float* obns = P(); const float* obnb = P();
    const float* ppn1c = P(); const float* ppn1s = P();
    const float* ppn2c = P(); const float* ppn2s = P();
    const float* ppn3c = P(); const float* ppn3p = P(); const float* ppn3sc = P();

    // ---- workspace bump allocator ----
    const int dim[5] = { 96, 48, 24, 12, 6 };
    const int C[5] = { 16, 32, 48, 64, 80 };
    long V[5]; for (int i = 0; i < 5; ++i) V[i] = (long)dim[i] * dim[i] * dim[i];
    char* wsp = (char*)d_ws; size_t off = 0;
    auto alloc = [&](size_t bytes) { void* p = wsp + off; off = (off + bytes + 255) & ~(size_t)255; return p; };

    float* mk[5]; for (int i = 0; i < 5; ++i) mk[i] = (float*)alloc(V[i] * 4);
    float*  grid  = (float*)alloc(V[0] * 4);
    half_t* gridh = (half_t*)alloc(V[0] * 16 * 2);   // 1ch expanded to 16ch
    float* skip[4]; for (int i = 0; i < 4; ++i) skip[i] = (float*)alloc(V[i] * C[i] * 4);
    float* fp2_0   = (float*)alloc(V[4] * 80 * 4);
    float* fp2_mid = (float*)alloc(V[2] * 48 * 4);
    float* xA   = (float*)alloc(V[0] * 32 * 4);
    float* xB   = (float*)alloc(V[0] * 32 * 4);
    float* tmp  = (float*)alloc(V[0] * 16 * 4);
    float* scb  = (float*)alloc(V[0] * 16 * 4);
    half_t* hbuf = (half_t*)alloc(V[0] * 32 * 2);
    half_t* hraw = (half_t*)alloc(V[0] * 32 * 2);
    float* stats = (float*)alloc((2 * 160 + 1) * 4);
    half_t* wtmp = (half_t*)alloc((size_t)27 * 5 * 5 * 512 * 2);
    float* att6  = (float*)alloc(V[4] * 4);
    float* att24 = (float*)alloc(V[2] * 4);
    float*  yA   = (float*)alloc(V[4] * 80 * 4);
    half_t* y4h  = (half_t*)alloc(V[4] * 80 * 2);
    half_t* ymh  = (half_t*)alloc(V[2] * 48 * 2);
    float*  yA2  = (float*)alloc(V[2] * 48 * 4);
    half_t* y2h  = (half_t*)alloc(V[2] * 48 * 2);
    half_t* zh   = (half_t*)alloc(V[0] * 16 * 2);
    float*  zA   = (float*)alloc(V[0] * 16 * 4);
    half_t* zch  = (half_t*)alloc(V[0] * 16 * 2);
    float*  pixb = (float*)alloc(V[0] * 3 * 4);
    float*  scob = (float*)alloc(V[0] * 2 * 4);
    (void)alloc(4096);                               // K-overread tail pad

    // ---- helpers ----
    auto conv = [&](const half_t* in, const float* w32, const float* mask,
                    const float* addsrc, float* out, half_t* outh,
                    int od, int idm, int CiA, int CiW, int Co, int mode) {
        int taps = (mode == 0) ? 27 : (mode == 3) ? 1 : 8;
        int KC = (CiA + 31) / 32;
        int CT = (Co + 15) / 16;
        long wn = (long)taps * KC * CT * 512;
        wrepack_k<<<gsz(wn), TPB, 0, stream>>>(w32, wtmp, taps, CiW, Co, KC, CT);
        int CTP = (CT + 1) / 2;
        int mt = (mode == 2) ? (((idm + 15) / 16) * 2) : ((od + 15) / 16);
        long nW = (long)od * od * mt * CTP;
        dim3 g((unsigned)((nW + 7) / 8));
        switch (mode) {
        case 0: conv_wmma<0><<<g, TPB, 0, stream>>>(in, wtmp, mask, addsrc, out, outh,
                    od, od, idm, idm, idm, CiA, Co, (int)nW); break;
        case 1: conv_wmma<1><<<g, TPB, 0, stream>>>(in, wtmp, mask, addsrc, out, outh,
                    od, od, idm, idm, idm, CiA, Co, (int)nW); break;
        case 2: conv_wmma<2><<<g, TPB, 0, stream>>>(in, wtmp, mask, addsrc, out, outh,
                    od, od, idm, idm, idm, CiA, Co, (int)nW); break;
        default: conv_wmma<3><<<g, TPB, 0, stream>>>(in, wtmp, mask, addsrc, out, outh,
                    od, od, idm, idm, idm, CiA, Co, (int)nW); break;
        }
    };
    auto bnrelu = [&](const float* f, const float* mask, const float* s,
                      const float* b, half_t* oh, long Vv, int Cc) {
        hipMemsetAsync(stats, 0, (2 * Cc + 1) * sizeof(float), stream);
        bn_stats<<<dim3(64, Cc), TPB, 0, stream>>>(f, mask, stats, Vv, Cc);
        bn_apply<<<gsz(Vv * Cc), TPB, 0, stream>>>(f, mask, stats, s, b, oh, Vv, Cc);
    };
    auto block = [&](const BlkP& p, const float* xin, float* xout,
                     const float* mask, int d, int Ci, int Co) {
        long Vv = (long)d * d * d;
        const float* scp = xin;
        if (p.nin) {
            cast_k<<<gsz(Vv * Ci), TPB, 0, stream>>>(xin, hraw, Vv * Ci);
            conv(hraw, p.nin, nullptr, nullptr, scb, nullptr, d, d, Ci, Ci, Co, 3);
            scp = scb;
        }
        bnrelu(xin, mask, p.bn1s, p.bn1b, hbuf, Vv, Ci);
        conv(hbuf, p.w1, mask, nullptr, tmp, nullptr, d, d, Ci, Ci, Co, 0);
        bnrelu(tmp, mask, p.bn2s, p.bn2b, hbuf, Vv, Co);
        conv(hbuf, p.w2, mask, scp, xout, nullptr, d, d, Co, Co, Co, 0);
    };

    // ---- voxelize + masks ----
    hipMemsetAsync(grid, 0, V[0] * 4, stream);
    hipMemsetAsync(mk[0], 0, V[0] * 4, stream);
    scatter_k<<<gsz(N), TPB, 0, stream>>>(pc, grid, mk[0], N);
    for (int i = 1; i < 5; ++i)
        pool_k<<<gsz(V[i]), TPB, 0, stream>>>(mk[i - 1], mk[i], dim[i]);

    // ---- input conv (1ch grid expanded to 16ch; weights padded in K) ----
    expand16_k<<<gsz(V[0] * 16), TPB, 0, stream>>>(grid, gridh, V[0]);
    conv(gridh, input_conv, mk[0], nullptr, xA, nullptr, 96, 96, 16, 1, 16, 0);

    float* xcur = xA; float* xalt = xB;
    auto swp = [&]() { float* t = xcur; xcur = xalt; xalt = t; };

    // ---- encoder ----
    for (int i = 0; i < 5; ++i) {
        for (int b = 0; b < 2; ++b) {
            block(enc[i].blk[b], xcur, xalt, mk[i], dim[i], C[i], C[i]); swp();
        }
        if (i < 4) {
            hipMemcpyAsync(skip[i], xcur, V[i] * C[i] * 4, hipMemcpyDeviceToDevice, stream);
            bnrelu(xcur, mk[i], enc[i].dbns, enc[i].dbnb, hbuf, V[i], C[i]);
            conv(hbuf, enc[i].dw, mk[i + 1], nullptr, xalt, nullptr,
                 dim[i + 1], dim[i], C[i], C[i], C[i + 1], 1);
            swp();
        }
    }
    hipMemcpyAsync(fp2_0, xcur, V[4] * 80 * 4, hipMemcpyDeviceToDevice, stream);

    // ---- decoder ----
    for (int i = 0; i < 4; ++i) {
        int L = 3 - i;
        bnrelu(xcur, mk[L + 1], dec[i].bns, dec[i].bnb, hbuf, V[L + 1], C[L + 1]);
        conv(hbuf, dec[i].dcw, mk[L], nullptr, tmp, nullptr,
             dim[L], dim[L + 1], C[L + 1], C[L + 1], C[L], 2);    // transpose conv
        concat_k<<<gsz(V[L] * 2 * C[L]), TPB, 0, stream>>>(skip[L], tmp, xalt, V[L], C[L]);
        swp();
        block(dec[i].blk[0], xcur, xalt, mk[L], dim[L], 2 * C[L], C[L]); swp();
        block(dec[i].blk[1], xcur, xalt, mk[L], dim[L], C[L], C[L]);     swp();
        if (L == 2)
            hipMemcpyAsync(fp2_mid, xcur, V[2] * 48 * 4, hipMemcpyDeviceToDevice, stream);
    }

    // ---- output offsets: (ppn3_out, seg, ppn1, ppn2, att, att2) ----
    long off_seg = (long)N * 5;
    long off_p1  = off_seg + (long)N * 5;
    long off_p2  = off_p1 + V[4] * 2;
    long off_att = off_p2 + V[2] * 2;
    long off_at2 = off_att + V[2];

    // ---- out_bn + seg head ----
    bnrelu(xcur, mk[0], obns, obnb, hbuf, V[0], 16);
    seg_k<<<gsz(N), TPB, 0, stream>>>(pc, hbuf, linw, linb, d_out + off_seg, N);

    // ---- PPN1 (6^3) ----
    cast_k<<<gsz(V[4] * 80), TPB, 0, stream>>>(fp2_0, hraw, V[4] * 80);
    conv(hraw, ppn1c, mk[4], nullptr, yA, y4h, 6, 6, 80, 80, 80, 0);
    conv(y4h, ppn1s, mk[4], nullptr, d_out + off_p1, nullptr, 6, 6, 80, 80, 2, 0);
    att_k<<<gsz(V[4]), TPB, 0, stream>>>(d_out + off_p1, mk[4], att6, (int)V[4]);
    upwrite_k<<<gsz(V[2]), TPB, 0, stream>>>(att6, d_out + off_att, 6, 24);

    // ---- PPN2 (24^3) ----
    mulatt_k<<<gsz(V[2] * 48), TPB, 0, stream>>>(fp2_mid, att6, ymh, 24, 6, 48);
    conv(ymh, ppn2c, mk[2], nullptr, yA2, y2h, 24, 24, 48, 48, 48, 0);
    conv(y2h, ppn2s, mk[2], nullptr, d_out + off_p2, nullptr, 24, 24, 48, 48, 2, 0);
    att_k<<<gsz(V[2]), TPB, 0, stream>>>(d_out + off_p2, mk[2], att24, (int)V[2]);
    upwrite_k<<<gsz(V[0]), TPB, 0, stream>>>(att24, d_out + off_at2, 24, 96);

    // ---- PPN3 (96^3) ----
    mulatt_k<<<gsz(V[0] * 16), TPB, 0, stream>>>(xcur, att24, zh, 96, 24, 16);
    conv(zh, ppn3c, mk[0], nullptr, zA, zch, 96, 96, 16, 16, 16, 0);
    conv(zch, ppn3p, mk[0], nullptr, pixb, nullptr, 96, 96, 16, 16, 3, 0);
    conv(zch, ppn3sc, mk[0], nullptr, scob, nullptr, 96, 96, 16, 16, 2, 0);
    p3gather_k<<<gsz(N), TPB, 0, stream>>>(pc, pixb, scob, d_out, N);

    (void)ws_size; (void)out_size;
}
